// TTAGN_20847771255352
// MI455X (gfx1250) — matrix operand order repaired and checked by reading, not compiled, not run on hardware
//
#include <hip/hip_runtime.h>
#include <hip/hip_bf16.h>
#include <math.h>

// ---------------------------------------------------------------------------
// Types for CDNA5 WMMA (wave32): V_WMMA_F32_16X16X32_BF16
//   A: 16x32 bf16 (8 VGPRs = v16bf), B: 32x16 bf16 (8 VGPRs = v16bf),
//   C/D: 16x16 f32 (8 VGPRs = v8f)
// ---------------------------------------------------------------------------
typedef __attribute__((ext_vector_type(16))) __bf16 v16bf;
typedef __attribute__((ext_vector_type(8)))  float  v8f;

// native f32 -> bf16 conversion (single v_cvt on gfx1250)
__device__ inline __bf16 f2bf(float f) { return (__bf16)f; }

// single-instruction v_rcp_f32 (approx is fine for activations/softmax)
__device__ inline float rcp_fast(float x) { return __builtin_amdgcn_rcpf(x); }

__device__ inline float sigf(float x) { return rcp_fast(1.0f + __expf(-x)); }

// branchless tanh: 1 - 2/(exp(2x)+1); saturates correctly for large |x|,
// avoids the divergent libm path and the IEEE-divide expansion
__device__ inline float tanh_fast(float x) {
    return 1.0f - 2.0f * rcp_fast(__expf(2.0f * x) + 1.0f);
}

__device__ inline void atomAddF(float* p, float v) { atomicAdd(p, v); }

// float atomic-max via int/uint monotonic-bit trick (no NaNs expected)
__device__ inline void atomMaxF(float* addr, float value) {
    if (value >= 0.0f) atomicMax((int*)addr, __float_as_int(value));
    else               atomicMin((unsigned int*)addr, __float_as_uint(value));
}

// ---------------------------------------------------------------------------
// LSTM over padded [B,T,2] sequences, hidden=16, returning h at step len-1.
// One wave = 16 sequences. Per step the gate GEMM is [16x32]@[32x64]:
//   K rows 0..1  = x_t  (Wih^T), K rows 2..17 = h (Whh^T), rows 18..31 = 0.
// Gate tiles (N chunks of 16) in torch order: 0=i, 1=f, 2=g, 3=o.
// h state round-trips through a per-wave LDS slab to translate WMMA
// C-layout -> A-layout between steps. The slab is private to the wave, so
// no workgroup barrier is needed: same-wave DS ops are in-order (ISA ch.11),
// only a compiler scheduling fence is required.
// ---------------------------------------------------------------------------
#define HID 16

__global__ void __launch_bounds__(256, 1)
lstm_wmma_kernel(const float* __restrict__ seq,
                 const int*   __restrict__ lens,
                 const float* __restrict__ Wih,   // [64,2]
                 const float* __restrict__ Whh,   // [64,16]
                 const float* __restrict__ bias,  // [64]
                 float*       __restrict__ feat,  // [total,16]
                 int total, int T) {
    __shared__ float hbuf[8][16][17];   // per-wave h slab (pad 17: bank-safe)
    const int lane = threadIdx.x & 31;
    const int wave = threadIdx.x >> 5;
    const int half = lane >> 4;         // 0: lanes 0-15, 1: lanes 16-31
    const int l16  = lane & 15;
    const int seqbase = blockIdx.x * 128 + wave * 16;

    // Build the 4 gate-tile B fragments (32x16 each) once.
    // B layout: lane -> N=l16 (halves give K blocks), element e -> K = e+16*half.
    v16bf Bf[4];
#pragma unroll
    for (int t = 0; t < 4; ++t) {
        int j = t * 16 + l16;                       // gate output index 0..63
#pragma unroll
        for (int e = 0; e < 16; ++e) {
            int k = e + 16 * half;                  // K row 0..31
            float w = 0.0f;
            if (k < 2)       w = Wih[j * 2 + k];
            else if (k < 18) w = Whh[j * 16 + (k - 2)];
            Bf[t][e] = f2bf(w);
        }
    }
    float bgate[4];
#pragma unroll
    for (int t = 0; t < 4; ++t) bgate[t] = bias[t * 16 + l16];

    // C-layout rows this lane owns (vgpr v -> row v + 8*half) + their lens.
    int rowv[8], lenv[8];
#pragma unroll
    for (int v = 0; v < 8; ++v) {
        int r = seqbase + v + 8 * half;
        rowv[v] = r;
        lenv[v] = lens[r < total ? r : (total - 1)];
    }
    const int arow  = seqbase + l16;                // A-layout row of this lane
    const int arowc = arow < total ? arow : (total - 1);

    v8f c, last;
#pragma unroll
    for (int v = 0; v < 8; ++v) { c[v] = 0.0f; last[v] = 0.0f; }
#pragma unroll
    for (int v = 0; v < 8; ++v) hbuf[wave][v + 8 * half][l16] = 0.0f;
    __builtin_amdgcn_wave_barrier();

    for (int t = 0; t < T; ++t) {
        // x_t for this lane's A-row
        float x0 = seq[(size_t)(arowc * T + t) * 2 + 0];
        float x1 = seq[(size_t)(arowc * T + t) * 2 + 1];
        // A layout: element e -> K = e + 16*half
        v16bf Af;
#pragma unroll
        for (int e = 0; e < 16; ++e) {
            int k = e + 16 * half;
            float vv = 0.0f;
            if (k < 2)       vv = (k == 0) ? x0 : x1;
            else if (k < 18) vv = hbuf[wave][l16][k - 2];
            Af[e] = f2bf(vv);
        }
        __builtin_amdgcn_wave_barrier();   // slab reads done before overwrite
        v8f g[4];
#pragma unroll
        for (int tg = 0; tg < 4; ++tg) {
            v8f accv;
#pragma unroll
            for (int v = 0; v < 8; ++v) accv[v] = bgate[tg];
            g[tg] = __builtin_amdgcn_wmma_f32_16x16x32_bf16(
                false, Af, false, Bf[tg], (short)0, accv, false, false);
        }
#pragma unroll
        for (int v = 0; v < 8; ++v) {
            float ig = sigf(g[0][v]);
            float fg = sigf(g[1][v]);
            float gg = tanh_fast(g[2][v]);
            float og = sigf(g[3][v]);
            float cv = fg * c[v] + ig * gg;
            float hv = og * tanh_fast(cv);
            c[v] = cv;
            if (t == lenv[v] - 1) last[v] = hv;
            hbuf[wave][v + 8 * half][l16] = hv;     // C-layout -> slab
        }
        __builtin_amdgcn_wave_barrier();   // slab writes before next-step reads
    }
#pragma unroll
    for (int v = 0; v < 8; ++v)
        if (rowv[v] < total) feat[(size_t)rowv[v] * HID + l16] = last[v];
}

// ---------------------------------------------------------------------------
// Generic WMMA GEMM: out[M,Nc] = act( A[M,K] @ W[K,Nc] + bias )
// Optional split input: cols [0,K1) from A (stride K1), [K1,K) from A2.
// bf16 inputs, f32 accumulation. One wave -> 16 output rows, loops N tiles.
// ---------------------------------------------------------------------------
__global__ void __launch_bounds__(256, 1)
gemm_wmma_kernel(const float* __restrict__ A,
                 const float* __restrict__ A2, int K1,
                 const float* __restrict__ W,
                 const float* __restrict__ bias,
                 float* __restrict__ out,
                 int M, int K, int Nc, int act) {
    const int lane = threadIdx.x & 31;
    const int wave = threadIdx.x >> 5;
    const int half = lane >> 4;
    const int l16  = lane & 15;
    const int rowbase = blockIdx.x * 128 + wave * 16;
    const int arow  = rowbase + l16;
    const int arowc = arow < M ? arow : (M - 1);
    const int ntiles = (Nc + 15) >> 4;
    const int K2 = K - K1;

    for (int nt = 0; nt < ntiles; ++nt) {
        int col = nt * 16 + l16;
        float bv = 0.0f;
        if (bias && col < Nc) bv = bias[col];
        v8f acc;
#pragma unroll
        for (int v = 0; v < 8; ++v) acc[v] = bv;

        for (int k0 = 0; k0 < K; k0 += 32) {
            v16bf Af, Bf;
#pragma unroll
            for (int e = 0; e < 16; ++e) {
                int ka = k0 + e + 16 * half;
                float av = 0.0f;
                if (ka < K)
                    av = (ka < K1) ? A [(size_t)arowc * K1 + ka]
                                   : A2[(size_t)arowc * K2 + (ka - K1)];
                Af[e] = f2bf(av);
                int kb = k0 + e + 16 * half;
                float wv = 0.0f;
                if (kb < K && col < Nc) wv = W[(size_t)kb * Nc + col];
                Bf[e] = f2bf(wv);
            }
            acc = __builtin_amdgcn_wmma_f32_16x16x32_bf16(
                false, Af, false, Bf, (short)0, acc, false, false);
        }
#pragma unroll
        for (int v = 0; v < 8; ++v) {
            int r = rowbase + v + 8 * half;
            if (r < M && col < Nc) {
                float o = acc[v];
                if (act == 1) o = o > 0.0f ? o : 0.01f * o;
                out[(size_t)r * Nc + col] = o;
            }
        }
    }
}

// ---------------------------------------------------------------------------
// Elementwise / graph kernels
// ---------------------------------------------------------------------------
__global__ void fill_kernel(float* p, float v, int n) {
    int i = blockIdx.x * blockDim.x + threadIdx.x;
    if (i < n) p[i] = v;
}

__global__ void dot16_bias_kernel(const float* __restrict__ feat,
                                  const float* __restrict__ w,
                                  const float* __restrict__ b,
                                  float* __restrict__ out, int n) {
    int i = blockIdx.x * blockDim.x + threadIdx.x;
    if (i >= n) return;
    float s = 0.0f;
#pragma unroll
    for (int j = 0; j < 16; ++j) s += feat[(size_t)i * 16 + j] * w[j];
    out[i] = s + b[0];
}

__global__ void concat_xin_kernel(const float* __restrict__ x,
                                  const float* __restrict__ nf,
                                  float* __restrict__ xin, int N, int D) {
    int i = blockIdx.x * blockDim.x + threadIdx.x;
    if (i >= N * 256) return;
    int n = i >> 8, j = i & 255;
    xin[i] = (j < D) ? x[(size_t)n * D + j] : nf[(size_t)n * 16 + (j - D)];
}

__global__ void gat_alpha_kernel(const float* __restrict__ xh,
                                 const float* __restrict__ att,
                                 float* __restrict__ alpha, int N, int H, int C) {
    int i = blockIdx.x * blockDim.x + threadIdx.x;
    if (i >= N * H) return;
    int n = i / H, h = i % H;
    float s = 0.0f;
    for (int c = 0; c < C; ++c) s += xh[(size_t)n * (H * C) + h * C + c] * att[h * C + c];
    alpha[i] = s;
}

__global__ void gat_pass1(const int* __restrict__ src, const int* __restrict__ dst,
                          const float* __restrict__ alpha,
                          float* __restrict__ abuf, float* __restrict__ amax,
                          int E, int H) {
    int i = blockIdx.x * blockDim.x + threadIdx.x;
    if (i >= E * H) return;
    int e = i / H, h = i % H;
    float a = alpha[src[e] * H + h];
    a = a > 0.0f ? a : 0.2f * a;                 // leaky_relu 0.2
    abuf[i] = a;
    atomMaxF(&amax[dst[e] * H + h], a);
}

__global__ void gat_pass2(const int* __restrict__ dst,
                          const float* __restrict__ amax,
                          float* __restrict__ abuf, float* __restrict__ esum,
                          int E, int H) {
    int i = blockIdx.x * blockDim.x + threadIdx.x;
    if (i >= E * H) return;
    int e = i / H, h = i % H;
    float ex = __expf(abuf[i] - amax[dst[e] * H + h]);
    abuf[i] = ex;
    atomAddF(&esum[dst[e] * H + h], ex);
}

__global__ void gat_init_acc_kernel(const float* __restrict__ xh,
                                    const float* __restrict__ bias,
                                    float* __restrict__ acc, int N, int F) {
    int i = blockIdx.x * blockDim.x + threadIdx.x;
    if (i >= N * F) return;
    acc[i] = xh[i] + bias[i % F];                // residual + bias
}

__global__ void gat_pass3(const int* __restrict__ src, const int* __restrict__ dst,
                          const float* __restrict__ abuf, const float* __restrict__ esum,
                          const float* __restrict__ escal, const float* __restrict__ xh,
                          float* __restrict__ acc, int E, int H, int C) {
    int i = blockIdx.x * blockDim.x + threadIdx.x;
    int F = H * C;
    if (i >= E * F) return;
    int e = i / F, j = i % F, h = j / C;
    int s = src[e], d = dst[e];
    float coef = abuf[e * H + h] * rcp_fast(esum[d * H + h]) + escal[e];
    atomAddF(&acc[(size_t)d * F + j], xh[(size_t)s * F + j] * coef);
}

__global__ void sanitize_kernel(const float* __restrict__ in,
                                float* __restrict__ out, int n) {
    int i = blockIdx.x * blockDim.x + threadIdx.x;
    if (i >= n) return;
    float v = in[i];
    out[i] = (__builtin_isnan(v) || __builtin_isinf(v)) ? 0.0f : v;
}

__global__ void deg_kernel(const int* __restrict__ dst, float* __restrict__ deg, int E) {
    int i = blockIdx.x * blockDim.x + threadIdx.x;
    if (i < E) atomAddF(&deg[dst[i]], 1.0f);
}

__global__ void rsqrt_kernel(float* __restrict__ p, int n) {
    int i = blockIdx.x * blockDim.x + threadIdx.x;
    if (i < n) p[i] = rsqrtf(p[i]);
}

__global__ void gcn_init_kernel(const float* __restrict__ xw,
                                const float* __restrict__ dinv,
                                const float* __restrict__ b,
                                float* __restrict__ acc, int N) {
    int i = blockIdx.x * blockDim.x + threadIdx.x;
    if (i >= N * 64) return;
    int n = i >> 6, j = i & 63;
    float di = dinv[n];
    acc[i] = xw[i] * di * di + b[j];
}

__global__ void gcn_scatter_kernel(const int* __restrict__ src, const int* __restrict__ dst,
                                   const float* __restrict__ xw,
                                   const float* __restrict__ dinv,
                                   float* __restrict__ acc, int E) {
    int i = blockIdx.x * blockDim.x + threadIdx.x;
    if (i >= E * 64) return;
    int e = i >> 6, c = i & 63;
    int s = src[e], d = dst[e];
    float norm = dinv[s] * dinv[d];
    atomAddF(&acc[(size_t)d * 64 + c], xw[(size_t)s * 64 + c] * norm);
}

__global__ void leaky_kernel(const float* __restrict__ in, float* __restrict__ out, int n) {
    int i = blockIdx.x * blockDim.x + threadIdx.x;
    if (i >= n) return;
    float v = in[i];
    out[i] = v > 0.0f ? v : 0.01f * v;
}

// ---------------------------------------------------------------------------
// Host-side orchestration
// ---------------------------------------------------------------------------
static inline int cdiv(long long a, int b) { return (int)((a + b - 1) / b); }

extern "C" void kernel_launch(void* const* d_in, const int* in_sizes, int n_in,
                              void* d_out, int out_size, void* d_ws, size_t ws_size,
                              hipStream_t stream) {
    const float* x     = (const float*)d_in[0];
    const float* eseq  = (const float*)d_in[1];
    const float* nseq  = (const float*)d_in[2];
    const float* Wih   = (const float*)d_in[3];
    const float* Whh   = (const float*)d_in[4];
    const float* lb    = (const float*)d_in[5];
    const float* dnnW  = (const float*)d_in[6];
    const float* dnnb  = (const float*)d_in[7];
    const float* g1W   = (const float*)d_in[8];
    const float* g1att = (const float*)d_in[9];
    const float* g1b   = (const float*)d_in[10];
    const float* g2W   = (const float*)d_in[11];
    const float* g2att = (const float*)d_in[12];
    const float* g2b   = (const float*)d_in[13];
    const float* gc1W  = (const float*)d_in[14];
    const float* gc1b  = (const float*)d_in[15];
    const float* gc2W  = (const float*)d_in[16];
    const float* gc2b  = (const float*)d_in[17];
    const float* c1W   = (const float*)d_in[18];
    const float* c1b   = (const float*)d_in[19];
    const float* c2W   = (const float*)d_in[20];
    const float* c2b   = (const float*)d_in[21];
    const float* c3W   = (const float*)d_in[22];
    const float* c3b   = (const float*)d_in[23];
    const int*   eidx  = (const int*)d_in[24];
    const int*   elen  = (const int*)d_in[25];
    const int*   nlen  = (const int*)d_in[26];

    const int E = in_sizes[25];
    const int N = in_sizes[26];
    const int T = in_sizes[1] / (E * 2);
    const int D = in_sizes[0] / N;              // 240; xin width = D+16 = 256

    const int* src = eidx;
    const int* dst = eidx + E;

    float* out0    = (float*)d_out;             // [N,4]
    float* out_gcn = out0 + (size_t)N * 4;      // [N,64]
    float* out_gat = out_gcn + (size_t)N * 64;  // [N,64]

    // workspace layout (floats); buffers are reused across phases
    float* ws = (float*)d_ws;
    size_t o = 0;
    float* f_escal = ws + o; o += (size_t)E;
    float* f_nfeat = ws + o; o += (size_t)N * 16;
    float* f_xin   = ws + o; o += (size_t)N * 256;
    float* f_efeat = ws + o; o += (size_t)E * 16;   // later: cls h2 [N,16]
    float* f_xh    = ws + o; o += (size_t)N * 64;   // gat1 xh -> gcn xw
    float* f_acc   = ws + o; o += (size_t)N * 64;   // gat1 out -> gcn1 acc
    float* f_xh2   = ws + o; o += (size_t)N * 64;   // gat2 xh -> gcn x1
    float* f_acc2  = ws + o; o += (size_t)N * 64;   // gat2 out -> gcn2 acc
    float* f_alpha = ws + o; o += (size_t)N * 8;
    float* f_amax  = ws + o; o += (size_t)N * 8;
    float* f_esum  = ws + o; o += (size_t)N * 8;
    float* f_abuf  = ws + o; o += (size_t)E * 8;    // later: cls h1 [N,32]
    float* f_deg   = ws + o; o += (size_t)N;        // deg -> dinv in place

    const int TB = 256;

    // ---- temporal branches (WMMA LSTM) ----
    lstm_wmma_kernel<<<cdiv(E, 128), TB, 0, stream>>>(eseq, elen, Wih, Whh, lb, f_efeat, E, T);
    lstm_wmma_kernel<<<cdiv(N, 128), TB, 0, stream>>>(nseq, nlen, Wih, Whh, lb, f_nfeat, N, T);
    dot16_bias_kernel<<<cdiv(E, TB), TB, 0, stream>>>(f_efeat, dnnW, dnnb, f_escal, E);
    concat_xin_kernel<<<cdiv((long long)N * 256, TB), TB, 0, stream>>>(x, f_nfeat, f_xin, N, D);

    // ---- GAT layer 1 (H=8, C=8) ----
    gemm_wmma_kernel<<<cdiv(N, 128), TB, 0, stream>>>(f_xin, nullptr, 256, g1W, nullptr, f_xh, N, 256, 64, 0);
    gat_alpha_kernel<<<cdiv((long long)N * 8, TB), TB, 0, stream>>>(f_xh, g1att, f_alpha, N, 8, 8);
    fill_kernel<<<cdiv((long long)N * 8, TB), TB, 0, stream>>>(f_amax, -1e30f, N * 8);
    fill_kernel<<<cdiv((long long)N * 8, TB), TB, 0, stream>>>(f_esum, 0.0f, N * 8);
    gat_pass1<<<cdiv((long long)E * 8, TB), TB, 0, stream>>>(src, dst, f_alpha, f_abuf, f_amax, E, 8);
    gat_pass2<<<cdiv((long long)E * 8, TB), TB, 0, stream>>>(dst, f_amax, f_abuf, f_esum, E, 8);
    gat_init_acc_kernel<<<cdiv((long long)N * 64, TB), TB, 0, stream>>>(f_xh, g1b, f_acc, N, 64);
    gat_pass3<<<cdiv((long long)E * 64, TB), TB, 0, stream>>>(src, dst, f_abuf, f_esum, f_escal, f_xh, f_acc, E, 8, 8);

    // ---- GAT layer 2 (H=1, C=64) ----
    gemm_wmma_kernel<<<cdiv(N, 128), TB, 0, stream>>>(f_acc, nullptr, 64, g2W, nullptr, f_xh2, N, 64, 64, 0);
    gat_alpha_kernel<<<cdiv(N, TB), TB, 0, stream>>>(f_xh2, g2att, f_alpha, N, 1, 64);
    fill_kernel<<<cdiv(N, TB), TB, 0, stream>>>(f_amax, -1e30f, N);
    fill_kernel<<<cdiv(N, TB), TB, 0, stream>>>(f_esum, 0.0f, N);
    gat_pass1<<<cdiv(E, TB), TB, 0, stream>>>(src, dst, f_alpha, f_abuf, f_amax, E, 1);
    gat_pass2<<<cdiv(E, TB), TB, 0, stream>>>(dst, f_amax, f_abuf, f_esum, E, 1);
    gat_init_acc_kernel<<<cdiv((long long)N * 64, TB), TB, 0, stream>>>(f_xh2, g2b, f_acc2, N, 64);
    gat_pass3<<<cdiv((long long)E * 64, TB), TB, 0, stream>>>(src, dst, f_abuf, f_esum, f_escal, f_xh2, f_acc2, E, 1, 64);
    sanitize_kernel<<<cdiv((long long)N * 64, TB), TB, 0, stream>>>(f_acc2, out_gat, N * 64);

    // ---- GCN stack ----
    fill_kernel<<<cdiv(N, TB), TB, 0, stream>>>(f_deg, 1.0f, N);
    deg_kernel<<<cdiv(E, TB), TB, 0, stream>>>(dst, f_deg, E);
    rsqrt_kernel<<<cdiv(N, TB), TB, 0, stream>>>(f_deg, N);   // now dinv

    gemm_wmma_kernel<<<cdiv(N, 128), TB, 0, stream>>>(f_xin, nullptr, 256, gc1W, nullptr, f_xh, N, 256, 64, 0);
    gcn_init_kernel<<<cdiv((long long)N * 64, TB), TB, 0, stream>>>(f_xh, f_deg, gc1b, f_acc, N);
    gcn_scatter_kernel<<<cdiv((long long)E * 64, TB), TB, 0, stream>>>(src, dst, f_xh, f_deg, f_acc, E);
    leaky_kernel<<<cdiv((long long)N * 64, TB), TB, 0, stream>>>(f_acc, f_xh2, N * 64);   // x1

    gemm_wmma_kernel<<<cdiv(N, 128), TB, 0, stream>>>(f_xh2, nullptr, 64, gc2W, nullptr, f_xh, N, 64, 64, 0);
    gcn_init_kernel<<<cdiv((long long)N * 64, TB), TB, 0, stream>>>(f_xh, f_deg, gc2b, f_acc2, N);
    gcn_scatter_kernel<<<cdiv((long long)E * 64, TB), TB, 0, stream>>>(src, dst, f_xh, f_deg, f_acc2, E);
    leaky_kernel<<<cdiv((long long)N * 64, TB), TB, 0, stream>>>(f_acc2, out_gcn, N * 64); // x_gcn

    // ---- classifier: h = [xin | x_gcn] (split-K WMMA GEMM) ----
    float* f_h1 = f_abuf;   // [N,32] fits in E*8 region
    float* f_h2 = f_efeat;  // [N,16] fits in E*16 region
    gemm_wmma_kernel<<<cdiv(N, 128), TB, 0, stream>>>(f_xin, out_gcn, 256, c1W, c1b, f_h1, N, 320, 32, 1);
    gemm_wmma_kernel<<<cdiv(N, 128), TB, 0, stream>>>(f_h1, nullptr, 32, c2W, c2b, f_h2, N, 32, 16, 1);
    gemm_wmma_kernel<<<cdiv(N, 128), TB, 0, stream>>>(f_h2, nullptr, 16, c3W, c3b, out0, N, 16, 4, 0);
}
